// Conv1d_57346403336652
// MI455X (gfx1250) — matrix operand order repaired and checked by reading, not compiled, not run on hardware
//
#include <hip/hip_runtime.h>

// Fused Conv1d-transpose(x2) + FIR([0.4,1.2,0.4]) + bias as a single GEMM:
//   Out[32768, 512] = im2col(x)[32768, 768] x Wcat[768, 512] + bias
// Split-bf16 scheme (hi/lo decomposition, 3x V_WMMA_F32_16X16X32_BF16 per K=32
// chunk, f32 accumulation): 2.67x fewer matrix instructions than f32 WMMA K=4,
// Even-output-column tiles skip the
// structurally-zero tap-2 K range (K=512 instead of 768): -16.7% matrix ops.

typedef __attribute__((ext_vector_type(4)))  float  v4f;
typedef __attribute__((ext_vector_type(8)))  float  v8f;
typedef __attribute__((ext_vector_type(4)))  __bf16 v4bf;
typedef __attribute__((ext_vector_type(8)))  __bf16 v8bf;
typedef __attribute__((ext_vector_type(16))) __bf16 v16bf;

#define H_IN   4096
#define C_INCH 256
#define NBATCH 8
#define M_TOT  (NBATCH * H_IN)   // 32768
#define K_TOT  768
#define N_TOT  512

#define BM 128
#define BN 128
#define BK 32
#define LDK 40   // padded bf16 K-stride: 80B rows -> 20-dword stride, conflict-free

__device__ __forceinline__ void split2(float x, __bf16& h, __bf16& l) {
    h = (__bf16)x;               // RNE to bf16
    l = (__bf16)(x - (float)h);  // residual (exact in f32), RNE to bf16
}

__device__ __forceinline__ v16bf cat16(v8bf a, v8bf b) {
    return __builtin_shufflevector(a, b, 0, 1, 2, 3, 4, 5, 6, 7,
                                         8, 9, 10, 11, 12, 13, 14, 15);
}

// ---------------------------------------------------------------------------
// Build the fused 768x512 weight, split into bf16 hi/lo, store TRANSPOSED
// (N-major, K-contiguous) so the GEMM can stage B with coalesced b128 copies.
__global__ void prep_wcat_split(const float* __restrict__ w,
                                __bf16* __restrict__ wtT_hi,
                                __bf16* __restrict__ wtT_lo) {
    int idx = blockIdx.x * blockDim.x + threadIdx.x;
    if (idx >= K_TOT * N_TOT) return;
    int row = idx >> 9;          // K index 0..767
    int col = idx & 511;         // N index 0..511
    int tap = row >> 8;          // 0..2
    int i   = row & 255;
    int o   = col & 255;
    const float fa = 0.4f, fc = 1.2f;   // FIR = (1,3,1)/5 * 2
    float w0 = w[(0 * 256 + i) * 256 + o];
    float w1 = w[(1 * 256 + i) * 256 + o];
    float w2 = w[(2 * 256 + i) * 256 + o];
    float v;
    if (col < 256) {  // even output position 2m: taps {m-1, m}; tap2 == 0
        v = (tap == 0) ? (fc * w0 + fa * w1)
          : (tap == 1) ? (fa * w1 + fc * w2)
          : 0.0f;
    } else {          // odd output position 2m+1: taps {m-1, m, m+1}
        v = (tap == 0) ? (fa * w0)
          : (tap == 1) ? (fa * w0 + fc * w1 + fa * w2)
          : (fa * w2);
    }
    __bf16 h, l;
    split2(v, h, l);
    wtT_hi[(size_t)col * K_TOT + row] = h;
    wtT_lo[(size_t)col * K_TOT + row] = l;
}

// ---------------------------------------------------------------------------
__global__ __launch_bounds__(256)
void fused_convt_fir_gemm(const float* __restrict__ x,
                          const __bf16* __restrict__ wtT_hi,
                          const __bf16* __restrict__ wtT_lo,
                          const float* __restrict__ bias,
                          float* __restrict__ out) {
    __shared__ __bf16 Ah[BM * LDK];   // A tile hi, row-major [m][k]
    __shared__ __bf16 Al[BM * LDK];   // A tile lo
    __shared__ __bf16 Bh[BN * LDK];   // B tile hi, col-major [n][k]
    __shared__ __bf16 Bl[BN * LDK];   // B tile lo

    const int nTile = blockIdx.x;     // 0..3
    const int mTile = blockIdx.y;     // 0..255
    const int r0 = mTile * BM;
    const int n0 = nTile * BN;

    const int tid  = threadIdx.x;
    const int wid  = tid >> 5;        // wave id 0..7 (wave32)
    const int lane = tid & 31;
    const int half = lane >> 4;
    const int lr   = lane & 15;
    const int wm   = wid & 1;         // wave M slab (64 rows)
    const int wn   = wid >> 1;        // wave N slab (32 cols)

    // Even-output columns (n0 < 256) have Wcat[tap=2][*] == 0: stop K at 512.
    const int kbEnd = (n0 < 256) ? (512 / BK) : (K_TOT / BK);

    v8f acc[4][2];
    #pragma unroll
    for (int mt = 0; mt < 4; ++mt)
        #pragma unroll
        for (int nt = 0; nt < 2; ++nt)
            acc[mt][nt] = v8f{};

    for (int kb = 0; kb < kbEnd; ++kb) {
        const int c0  = kb * BK;
        const int tap = c0 >> 8;      // each 32-wide K slice lies in one tap
        const int i0  = c0 & 255;

        // ---- stage A: load f32 x rows, split to bf16 hi/lo on the fly
        {
            const int cc = (tid & 7) * 4;           // 0..28
            #pragma unroll
            for (int p = 0; p < 4; ++p) {
                const int rr = (tid >> 3) + p * 32; // 0..127
                const int r  = r0 + rr;
                const int h  = (r & (H_IN - 1)) + tap - 1;   // row within batch
                v4f v = v4f{};
                if ((unsigned)h < (unsigned)H_IN)
                    v = *(const v4f*)(x + (size_t)(r + tap - 1) * C_INCH + i0 + cc);
                v4bf vh, vl;
                #pragma unroll
                for (int j = 0; j < 4; ++j) {
                    __bf16 bhh, bll;
                    split2(v[j], bhh, bll);
                    vh[j] = bhh;
                    vl[j] = bll;
                }
                *(v4bf*)&Ah[rr * LDK + cc] = vh;
                *(v4bf*)&Al[rr * LDK + cc] = vl;
            }
        }
        // ---- stage B: pre-split bf16 weights, K-contiguous -> b128 copies
        {
            const int kc = (tid & 3) * 8;           // 0..24
            #pragma unroll
            for (int p = 0; p < 2; ++p) {
                const int col = (tid >> 2) + p * 64;            // 0..127
                const size_t src = (size_t)(n0 + col) * K_TOT + c0 + kc;
                *(v8bf*)&Bh[col * LDK + kc] = *(const v8bf*)&wtT_hi[src];
                *(v8bf*)&Bl[col * LDK + kc] = *(const v8bf*)&wtT_lo[src];
            }
        }
        __syncthreads();

        // ---- fragments: documented 16-bit A/B wave32 layouts
        // A lane<16: K0-7 (v0-3), K16-23 (v4-7); lane>=16: K8-15, K24-31
        // B lane<16: K0-7 (v0-3), K16-23 (v4-7); lane>=16: K8-15, K24-31; column = lane&15
        v16bf bh[2], bl[2];
        #pragma unroll
        for (int nt = 0; nt < 2; ++nt) {
            const int nrow = (wn * 32 + nt * 16 + lr) * LDK + 8 * half;
            bh[nt] = cat16(*(const v8bf*)&Bh[nrow], *(const v8bf*)&Bh[nrow + 16]);
            bl[nt] = cat16(*(const v8bf*)&Bl[nrow], *(const v8bf*)&Bl[nrow + 16]);
        }
        #pragma unroll
        for (int mt = 0; mt < 4; ++mt) {
            const int arow = (wm * 64 + mt * 16 + lr) * LDK + 8 * half;
            v16bf ah = cat16(*(const v8bf*)&Ah[arow], *(const v8bf*)&Ah[arow + 16]);
            v16bf al = cat16(*(const v8bf*)&Al[arow], *(const v8bf*)&Al[arow + 16]);
            #pragma unroll
            for (int nt = 0; nt < 2; ++nt) {
                // x*w ~= hi*hi + hi*lo + lo*hi  (lo*lo ~ 2^-16, dropped)
                acc[mt][nt] = __builtin_amdgcn_wmma_f32_16x16x32_bf16(
                    false, ah, false, bh[nt], (short)0, acc[mt][nt], false, false);
                acc[mt][nt] = __builtin_amdgcn_wmma_f32_16x16x32_bf16(
                    false, ah, false, bl[nt], (short)0, acc[mt][nt], false, false);
                acc[mt][nt] = __builtin_amdgcn_wmma_f32_16x16x32_bf16(
                    false, al, false, bh[nt], (short)0, acc[mt][nt], false, false);
            }
        }
        __syncthreads();
    }

    // ---- epilogue: C/D layout: VGPR j -> M = j + 8*half, N = lane&15
    const int colBaseWave = n0 + wn * 32;
    #pragma unroll
    for (int mt = 0; mt < 4; ++mt) {
        const int rowBase = r0 + wm * 64 + mt * 16 + half * 8;
        #pragma unroll
        for (int nt = 0; nt < 2; ++nt) {
            const int col = colBaseWave + nt * 16 + lr;
            const float bv = bias[col & 255];   // cols 256..511 = odd rows, same bias
            float* dst = out + (size_t)rowBase * N_TOT + col;
            #pragma unroll
            for (int j = 0; j < 8; ++j)
                dst[(size_t)j * N_TOT] = acc[mt][nt][j] + bv;
        }
    }
}

// ---------------------------------------------------------------------------
extern "C" void kernel_launch(void* const* d_in, const int* in_sizes, int n_in,
                              void* d_out, int out_size, void* d_ws, size_t ws_size,
                              hipStream_t stream) {
    const float* x = (const float*)d_in[0];   // [8, 4096, 256]
    const float* w = (const float*)d_in[1];   // [3, 256, 256]
    const float* b = (const float*)d_in[2];   // [256]
    float* out = (float*)d_out;               // [8, 8192, 256] == [32768, 512]

    // ws layout: wcatT_hi [512*768] bf16, wcatT_lo [512*768] bf16  (~1.6 MB)
    __bf16* wtT_hi = (__bf16*)d_ws;
    __bf16* wtT_lo = wtT_hi + (size_t)N_TOT * K_TOT;

    prep_wcat_split<<<dim3((K_TOT * N_TOT + 255) / 256), dim3(256), 0, stream>>>(
        w, wtT_hi, wtT_lo);

    dim3 grid(N_TOT / BN, M_TOT / BM);        // (4, 256)
    fused_convt_fir_gemm<<<grid, dim3(256), 0, stream>>>(x, wtT_hi, wtT_lo, b, out);
}
